// QuantizableMoEBlock_87342454931495
// MI455X (gfx1250) — matrix operand order repaired and checked by reading, not compiled, not run on hardware
//
#include <hip/hip_runtime.h>
#include <hip/hip_bf16.h>

#define NUM_TOKENS 1024
#define HID        2048
#define INTERD     1408
#define NEXP       8

typedef __attribute__((ext_vector_type(16))) __bf16       v16bf;
typedef __attribute__((ext_vector_type(8)))  float        v8f;
typedef __attribute__((ext_vector_type(4)))  float        f32x4;
typedef __attribute__((ext_vector_type(4)))  unsigned int u32x4;

union FragAB { u32x4 q[2]; v16bf v; };

#if defined(__has_builtin)
#if __has_builtin(__builtin_amdgcn_cvt_pk_bf16_f32)
#define HAVE_CVT_PK_BF16 1
#endif
#endif

// Pack two f32 -> packed bf16x2 (lo in [15:0]). Round-to-nearest; on
// toolchains with v_cvt_pk_bf16_f32 use it, else 2x v_add + 1x v_perm_b32.
__device__ __forceinline__ unsigned bfpack(float lo, float hi) {
#ifdef HAVE_CVT_PK_BF16
  auto rr = __builtin_amdgcn_cvt_pk_bf16_f32(lo, hi);
  unsigned u;
  __builtin_memcpy(&u, &rr, sizeof(unsigned));
  return u;
#else
  unsigned a = __float_as_uint(lo) + 0x8000u;   // round-to-nearest (ties away)
  unsigned b = __float_as_uint(hi) + 0x8000u;
  // {b:a} bytes: sel 2,3 -> a.hi16 ; sel 6,7 -> b.hi16
  return __builtin_amdgcn_perm(b, a, 0x07060302u);
#endif
}
__device__ __forceinline__ unsigned short bf16of(float f) {
  return (unsigned short)((__float_as_uint(f) + 0x8000u) >> 16);
}
__device__ __forceinline__ u32x4 packq(f32x4 x, f32x4 y) {
  u32x4 q = { bfpack(x.x, x.y), bfpack(x.z, x.w), bfpack(y.x, y.y), bfpack(y.z, y.w) };
  return q;
}

// ---------------------------------------------------------------------------
// Kernel 1: router logits -> softmax -> top-2 -> renormalized combine[1024][8]
// fp32 throughout (top-k selection must match the fp32 reference).
// ---------------------------------------------------------------------------
__global__ __launch_bounds__(256) void moe_router(
    const float* __restrict__ H, const float* __restrict__ R,
    float* __restrict__ comb)
{
  const int t = blockIdx.x, tid = threadIdx.x;
  __shared__ float red[8 * 256];

  const int k = tid * 8;                         // 256 threads * 8 = 2048
  f32x4 h0 = *(const f32x4*)(H + (size_t)t * HID + k);
  f32x4 h1 = *(const f32x4*)(H + (size_t)t * HID + k + 4);
#pragma unroll
  for (int e = 0; e < 8; ++e) {
    f32x4 r0 = *(const f32x4*)(R + (size_t)e * HID + k);
    f32x4 r1 = *(const f32x4*)(R + (size_t)e * HID + k + 4);
    f32x4 p  = h0 * r0 + h1 * r1;
    red[e * 256 + tid] = p.x + p.y + p.z + p.w;
  }
  __syncthreads();
  for (int s = 128; s > 0; s >>= 1) {
    if (tid < s) {
#pragma unroll
      for (int e = 0; e < 8; ++e)
        red[e * 256 + tid] += red[e * 256 + tid + s];
    }
    __syncthreads();
  }
  if (tid == 0) {
    float l[8];
#pragma unroll
    for (int e = 0; e < 8; ++e) l[e] = red[e * 256];
    int i1 = 0; float v1 = l[0];
#pragma unroll
    for (int e = 1; e < 8; ++e) if (l[e] > v1) { v1 = l[e]; i1 = e; }
    int i2 = -1; float v2 = -3.4e38f;
#pragma unroll
    for (int e = 0; e < 8; ++e) if (e != i1 && l[e] > v2) { v2 = l[e]; i2 = e; }
    float p2 = __expf(v2 - v1);
    float w1 = 1.0f / (1.0f + p2);
    float w2 = p2 * w1;
#pragma unroll
    for (int e = 0; e < 8; ++e) {
      float v = (e == i1) ? w1 : ((e == i2) ? w2 : 0.0f);
      comb[(size_t)t * NEXP + e] = v;
    }
  }
}

// ============================ GEMM-1 helpers ===============================
struct StageRegs2 { f32x4 a0,a1,a2,a3, g0,g1,g2,g3, u0,u1,u2,u3; };

__device__ __forceinline__ void load_tile2(StageRegs2& s, const float* pH,
                                           const float* pG, const float* pU, int k) {
  const f32x4* pa = (const f32x4*)(pH + k);
  s.a0 = pa[0]; s.a1 = pa[1]; s.a2 = pa[2]; s.a3 = pa[3];
  const f32x4* pg = (const f32x4*)(pG + k);
  s.g0 = pg[0]; s.g1 = pg[1]; s.g2 = pg[2]; s.g3 = pg[3];
  const f32x4* pu = (const f32x4*)(pU + k);
  s.u0 = pu[0]; s.u1 = pu[1]; s.u2 = pu[2]; s.u3 = pu[3];
}
__device__ __forceinline__ void store_tile2(const StageRegs2& s, unsigned char* base,
                                            int r, int kc, float scG, float scU) {
  unsigned short* sA  = (unsigned short*)base;
  unsigned short* sBg = (unsigned short*)(base + 10240);
  unsigned short* sBu = (unsigned short*)(base + 20480);
  *(u32x4*)(sA  + r*40 + kc)     = packq(s.a0, s.a1);
  *(u32x4*)(sA  + r*40 + kc + 8) = packq(s.a2, s.a3);
  *(u32x4*)(sBg + r*40 + kc)     = packq(s.g0*scG, s.g1*scG);
  *(u32x4*)(sBg + r*40 + kc + 8) = packq(s.g2*scG, s.g3*scG);
  *(u32x4*)(sBu + r*40 + kc)     = packq(s.u0*scU, s.u1*scU);
  *(u32x4*)(sBu + r*40 + kc + 8) = packq(s.u2*scU, s.u3*scU);
}
__device__ __forceinline__ void compute_tile2(const unsigned char* base,
    int wm, int wn, int hf, int l15, v8f (&accG)[2][4], v8f (&accU)[2][4]) {
  const unsigned short* sA  = (const unsigned short*)base;
  const unsigned short* sBg = (const unsigned short*)(base + 10240);
  const unsigned short* sBu = (const unsigned short*)(base + 20480);
  FragAB a[2];
#pragma unroll
  for (int mt = 0; mt < 2; ++mt) {
    const unsigned short* p = sA + (wm*32 + mt*16 + l15)*40 + 16*hf;  // K=16h..16h+15
    a[mt].q[0] = *(const u32x4*)p;
    a[mt].q[1] = *(const u32x4*)(p + 8);
  }
#pragma unroll
  for (int nt = 0; nt < 4; ++nt) {
    int nrow = wn*64 + nt*16 + l15;
    const unsigned short* pg = sBg + nrow*40 + 16*hf;                // K=16h..16h+15
    FragAB bgf; bgf.q[0] = *(const u32x4*)pg; bgf.q[1] = *(const u32x4*)(pg + 8);
#pragma unroll
    for (int mt = 0; mt < 2; ++mt)
      accG[mt][nt] = __builtin_amdgcn_wmma_f32_16x16x32_bf16(
          false, a[mt].v, false, bgf.v, (short)0, accG[mt][nt], false, false);
    const unsigned short* pu = sBu + nrow*40 + 16*hf;
    FragAB buf2; buf2.q[0] = *(const u32x4*)pu; buf2.q[1] = *(const u32x4*)(pu + 8);
#pragma unroll
    for (int mt = 0; mt < 2; ++mt)
      accU[mt][nt] = __builtin_amdgcn_wmma_f32_16x16x32_bf16(
          false, a[mt].v, false, buf2.v, (short)0, accU[mt][nt], false, false);
  }
}

// ---------------------------------------------------------------------------
// Kernel 2: per expert gate/up GEMM (bf16 WMMA, fp32 acc). (1+scale) folded
// into the f32->bf16 conversion of B, bias in epilogue, then
// h'[e,t,i] = combine[t,e] * silu(gate) * up  stored bf16.
// 128x128 block tile, K=32 steps, double-buffered LDS, 1 barrier/step.
// ---------------------------------------------------------------------------
__global__ __launch_bounds__(256) void moe_gemm1(
    const float* __restrict__ H,  const float* __restrict__ W1,
    const float* __restrict__ sg, const float* __restrict__ bg,
    const float* __restrict__ su, const float* __restrict__ bu,
    const float* __restrict__ comb, unsigned short* __restrict__ hout)
{
  const int tid = threadIdx.x;
  const int n0  = blockIdx.x * 128;
  const int m0  = blockIdx.y * 128;
  const int e   = blockIdx.z;

  __shared__ __align__(16) unsigned char smem[62976];
  unsigned short* sO = (unsigned short*)smem;              // epilogue reuse: 128x128, stride 136
  float* pBg = (float*)(smem + 61440);
  float* pBu = (float*)(smem + 61952);
  float* pC  = (float*)(smem + 62464);

  if (tid < 128) {
    int i = n0 + tid;
    pBg[tid] = bg[(size_t)e * INTERD + i];
    pBu[tid] = bu[(size_t)e * INTERD + i];
    pC[tid]  = comb[(size_t)(m0 + tid) * NEXP + e];
  }

  const int lane = tid & 31, w = tid >> 5;
  const int wm = w >> 1, wn = w & 1;                 // 4 x 2 wave grid
  const int hf = lane >> 4, l15 = lane & 15;

  const int r  = tid >> 1;
  const int kc = (tid & 1) << 4;
  const float* pHrow = H + (size_t)(m0 + r) * HID + kc;
  const float* pGrow = W1 + ((size_t)e * (2*INTERD) + (n0 + r)) * HID + kc;
  const float* pUrow = pGrow + (size_t)INTERD * HID;
  const float scG = 1.0f + sg[(size_t)e * INTERD + n0 + r];
  const float scU = 1.0f + su[(size_t)e * INTERD + n0 + r];

  const v8f vzero = {0.f,0.f,0.f,0.f,0.f,0.f,0.f,0.f};
  v8f accG[2][4], accU[2][4];
#pragma unroll
  for (int mt = 0; mt < 2; ++mt)
#pragma unroll
    for (int nt = 0; nt < 4; ++nt) { accG[mt][nt] = vzero; accU[mt][nt] = vzero; }

  StageRegs2 sreg;
  load_tile2(sreg, pHrow, pGrow, pUrow, 0);
  store_tile2(sreg, smem, r, kc, scG, scU);
  __syncthreads();

  for (int i = 0; i < HID / 32; ++i) {
    const int cur = i & 1;
    if (i + 1 < HID / 32) load_tile2(sreg, pHrow, pGrow, pUrow, (i + 1) * 32);
    compute_tile2(smem + cur * 30720, wm, wn, hf, l15, accG, accU);
    if (i + 1 < HID / 32) store_tile2(sreg, smem + (cur ^ 1) * 30720, r, kc, scG, scU);
    __syncthreads();
  }

  // epilogue: bias, silu(gate)*up, * combine -> bf16 staged in LDS
#pragma unroll
  for (int mt = 0; mt < 2; ++mt)
#pragma unroll
    for (int nt = 0; nt < 4; ++nt) {
      int nloc = wn*64 + nt*16 + l15;
      float bgv = pBg[nloc], buv = pBu[nloc];
      v8f g = accG[mt][nt], u = accU[mt][nt];
#pragma unroll
      for (int v = 0; v < 8; ++v) {
        int mloc = wm*32 + mt*16 + 8*hf + v;          // C layout: M = v + 8*(lane>>4)
        float gate = g[v] + bgv;
        float up   = u[v] + buv;
        float hval = gate * (1.0f / (1.0f + __expf(-gate))) * up * pC[mloc];
        sO[mloc * 136 + nloc] = bf16of(hval);
      }
    }
  __syncthreads();
  { // coalesced 128B-per-thread store of the bf16 tile
    const int hc = tid & 1;
    u32x4* dst = (u32x4*)(hout + ((size_t)e * NUM_TOKENS + m0 + r) * INTERD + n0 + hc * 64);
    const unsigned short* srcp = sO + r * 136 + hc * 64;
#pragma unroll
    for (int qi = 0; qi < 8; ++qi) dst[qi] = *(const u32x4*)(srcp + qi * 8);
  }
}

// ============================ GEMM-2 helpers ===============================
struct StageRegs3 { u32x4 a0, a1; f32x4 b0, b1, b2, b3; float sc; };

__device__ __forceinline__ void load_tile3(StageRegs3& s,
    const unsigned short* hin, const float* Wd, const float* sd,
    int e, int k, int m0, int n0, int r, int kc) {
  const u32x4* pa = (const u32x4*)(hin + ((size_t)e * NUM_TOKENS + m0 + r) * INTERD + k + kc);
  s.a0 = pa[0]; s.a1 = pa[1];
  const f32x4* pb = (const f32x4*)(Wd + ((size_t)e * HID + n0 + r) * INTERD + k + kc);
  s.b0 = pb[0]; s.b1 = pb[1]; s.b2 = pb[2]; s.b3 = pb[3];
  s.sc = 1.0f + sd[(size_t)e * HID + n0 + r];
}
__device__ __forceinline__ void store_tile3(const StageRegs3& s, unsigned char* base,
                                            int r, int kc) {
  unsigned short* sA = (unsigned short*)base;
  unsigned short* sB = (unsigned short*)(base + 10240);
  *(u32x4*)(sA + r*40 + kc)     = s.a0;
  *(u32x4*)(sA + r*40 + kc + 8) = s.a1;
  *(u32x4*)(sB + r*40 + kc)     = packq(s.b0*s.sc, s.b1*s.sc);
  *(u32x4*)(sB + r*40 + kc + 8) = packq(s.b2*s.sc, s.b3*s.sc);
}
__device__ __forceinline__ void compute_tile3(const unsigned char* base,
    int wm, int wn, int hf, int l15, v8f (&acc)[2][4]) {
  const unsigned short* sA = (const unsigned short*)base;
  const unsigned short* sB = (const unsigned short*)(base + 10240);
  FragAB a[2];
#pragma unroll
  for (int mt = 0; mt < 2; ++mt) {
    const unsigned short* p = sA + (wm*32 + mt*16 + l15)*40 + 16*hf;
    a[mt].q[0] = *(const u32x4*)p;
    a[mt].q[1] = *(const u32x4*)(p + 8);
  }
#pragma unroll
  for (int nt = 0; nt < 4; ++nt) {
    const unsigned short* pb = sB + (wn*64 + nt*16 + l15)*40 + 16*hf;
    FragAB bf; bf.q[0] = *(const u32x4*)pb; bf.q[1] = *(const u32x4*)(pb + 8);
#pragma unroll
    for (int mt = 0; mt < 2; ++mt)
      acc[mt][nt] = __builtin_amdgcn_wmma_f32_16x16x32_bf16(
          false, a[mt].v, false, bf.v, (short)0, acc[mt][nt], false, false);
  }
}

// ---------------------------------------------------------------------------
// Kernel 3: down-proj. out[t,n] = sum_e (1+sd[e,n]) * (h'[e,t,:] . Wd[e,n,:])
//                                + sum_e comb[t,e]*bd[e,n]
// combine already folded into h'; sd folded into B conversion. Expert and K
// loops flattened into one 352-step double-buffered pipeline (no atomics).
// ---------------------------------------------------------------------------
__global__ __launch_bounds__(256) void moe_gemm2(
    const unsigned short* __restrict__ hin, const float* __restrict__ Wd,
    const float* __restrict__ sd, const float* __restrict__ bd,
    const float* __restrict__ comb, float* __restrict__ out)
{
  const int tid = threadIdx.x;
  const int n0  = blockIdx.x * 128;
  const int m0  = blockIdx.y * 128;

  __shared__ __align__(16) unsigned char smem[49152];
  float* pComb = (float*)(smem + 40960);                 // [128 tokens][8 experts]
  float* pBias = (float*)(smem + 45056);                 // [8 experts][128 n]

  for (int i = tid; i < 1024; i += 256) {
    pComb[i] = comb[(size_t)(m0 + (i >> 3)) * NEXP + (i & 7)];
    pBias[i] = bd[(size_t)(i >> 7) * HID + n0 + (i & 127)];
  }

  const int lane = tid & 31, w = tid >> 5;
  const int wm = w >> 1, wn = w & 1;
  const int hf = lane >> 4, l15 = lane & 15;
  const int r  = tid >> 1;
  const int kc = (tid & 1) << 4;

  const v8f vzero = {0.f,0.f,0.f,0.f,0.f,0.f,0.f,0.f};
  v8f acc[2][4];
#pragma unroll
  for (int mt = 0; mt < 2; ++mt)
#pragma unroll
    for (int nt = 0; nt < 4; ++nt) acc[mt][nt] = vzero;

  const int KT  = INTERD / 32;      // 44
  const int NIT = NEXP * KT;        // 352

  StageRegs3 sreg;
  load_tile3(sreg, hin, Wd, sd, 0, 0, m0, n0, r, kc);
  store_tile3(sreg, smem, r, kc);
  __syncthreads();

  for (int i = 0; i < NIT; ++i) {
    const int cur = i & 1;
    if (i + 1 < NIT) {
      int e2 = (i + 1) / KT;
      int k2 = ((i + 1) - e2 * KT) * 32;
      load_tile3(sreg, hin, Wd, sd, e2, k2, m0, n0, r, kc);
    }
    compute_tile3(smem + cur * 20480, wm, wn, hf, l15, acc);
    if (i + 1 < NIT) store_tile3(sreg, smem + (cur ^ 1) * 20480, r, kc);
    __syncthreads();
  }

  // epilogue: add sum_e comb[t,e]*bd[e,n], write fp32 out
#pragma unroll
  for (int mt = 0; mt < 2; ++mt)
#pragma unroll
    for (int nt = 0; nt < 4; ++nt) {
      int nloc = wn*64 + nt*16 + l15;
      v8f c = acc[mt][nt];
#pragma unroll
      for (int v = 0; v < 8; ++v) {
        int mloc = wm*32 + mt*16 + 8*hf + v;
        float b = 0.0f;
#pragma unroll
        for (int e2 = 0; e2 < 8; ++e2)
          b += pComb[mloc * 8 + e2] * pBias[e2 * 128 + nloc];
        out[(size_t)(m0 + mloc) * HID + n0 + nloc] = c[v] + b;
      }
    }
}

// ---------------------------------------------------------------------------
extern "C" void kernel_launch(void* const* d_in, const int* in_sizes, int n_in,
                              void* d_out, int out_size, void* d_ws, size_t ws_size,
                              hipStream_t stream) {
  (void)in_sizes; (void)n_in; (void)out_size; (void)ws_size;
  const float* H  = (const float*)d_in[0];
  const float* R  = (const float*)d_in[1];
  const float* W1 = (const float*)d_in[2];
  const float* Wd = (const float*)d_in[3];
  const float* sg = (const float*)d_in[4];
  const float* bg = (const float*)d_in[5];
  const float* su = (const float*)d_in[6];
  const float* bu = (const float*)d_in[7];
  const float* sd = (const float*)d_in[8];
  const float* bd = (const float*)d_in[9];
  float* out = (float*)d_out;

  float*          combine = (float*)d_ws;                              // 1024*8 f32 = 32 KB
  unsigned short* hbuf    = (unsigned short*)((char*)d_ws + 32768);    // 8*1024*1408 bf16 = 23 MB

  moe_router<<<NUM_TOKENS, 256, 0, stream>>>(H, R, combine);
  moe_gemm1<<<dim3(INTERD / 128, NUM_TOKENS / 128, NEXP), 256, 0, stream>>>(
      H, W1, sg, bg, su, bu, combine, hbuf);
  moe_gemm2<<<dim3(HID / 128, NUM_TOKENS / 128), 256, 0, stream>>>(
      hbuf, Wd, sd, bd, combine, out);
}
